// DreamModel_17420387352607
// MI455X (gfx1250) — compile-verified, matrix operand order checked by reading, not run on hardware
//
#include <hip/hip_runtime.h>

#define BDIM   512
#define TDIM   64
#define NBAG   16
#define EMB    256
#define GATES  768      // 3*EMB
#define NLAY   2
#define APAD   264      // A-tile LDS row stride in halves (16B-aligned, conflict-free)

typedef _Float16 h8   __attribute__((ext_vector_type(8)));
typedef _Float16 v16h __attribute__((ext_vector_type(16)));
typedef float    v8f  __attribute__((ext_vector_type(8)));

static __device__ __forceinline__ v16h frag_cat(h8 lo, h8 hi) {
  return __builtin_shufflevector(lo, hi, 0,1,2,3,4,5,6,7,8,9,10,11,12,13,14,15);
}
static __device__ __forceinline__ float sigmoidf_(float x) {
  return 1.0f / (1.0f + __expf(-x));
}

// Async-stage a 16x256-half A tile (contiguous rows in global) into LDS with
// row stride APAD. 256 threads x 2 chunks of 16B = 8 KB. gfx1250 async path:
// global_load_async_to_lds_b128 tracked by ASYNCcnt.
static __device__ __forceinline__ void stage_A_async(_Float16* aSh,
                                                     const _Float16* gsrc,
                                                     int tid) {
#pragma unroll
  for (int it = 0; it < 2; ++it) {
    int c   = tid + it * 256;            // 512 chunks of 16B
    int row = c >> 5;                    // 32 chunks per 512B row
    int col = c & 31;
    unsigned lds = (unsigned)(size_t)(aSh + row * APAD + col * 8);
    unsigned long long g = (unsigned long long)(size_t)(gsrc + row * EMB + col * 8);
    asm volatile("global_load_async_to_lds_b128 %0, %1, off"
                 :: "v"(lds), "v"(g) : "memory");
  }
  asm volatile("s_wait_asynccnt 0x0" ::: "memory");
}

// ---------------------------------------------------------------------------
__global__ void wconv_kernel(const float* __restrict__ wih, const float* __restrict__ whh,
                             _Float16* __restrict__ wih16, _Float16* __restrict__ whh16, int n) {
  int i = blockIdx.x * blockDim.x + threadIdx.x;
  if (i < n) { wih16[i] = (_Float16)wih[i]; whh16[i] = (_Float16)whh[i]; }
}

// ---------------------------------------------------------------------------
// Embedding gather + mean over bag of 16 -> f16 pooled (B*T, 256).
// Table (102 MB) is L2-resident on MI455X (192 MB L2).
__global__ __launch_bounds__(EMB) void pool_kernel(const int* __restrict__ x,
                                                   const float* __restrict__ emb,
                                                   _Float16* __restrict__ pooled) {
  __shared__ int idx[NBAG];
  int bt = blockIdx.x;          // bt = b*T + t
  int c  = threadIdx.x;
  if (c < NBAG) idx[c] = x[(long)bt * NBAG + c];
  __syncthreads();
  float s = 0.0f;
#pragma unroll
  for (int i = 0; i < NBAG; ++i) {
    int id = idx[i];
    if (id != 0) s += emb[(long)id * EMB + c];   // emb row 0 forced to zero
  }
  pooled[(long)bt * EMB + c] = (_Float16)(s * (1.0f / NBAG));
}

// ---------------------------------------------------------------------------
// Time-parallel input GEMM: gi = act(B*T x 256) @ W_ih^T(768x256) + b_ih -> f16.
// One block per 16-row M-tile: A staged once via async->LDS, 8 waves x 6 N-tiles
// cover all 768 columns. B fragments batched per K-chunk so wmma overlaps VMEM.
__global__ __launch_bounds__(256) void giGemm_kernel(const _Float16* __restrict__ act,
                                                     const _Float16* __restrict__ w,
                                                     const float* __restrict__ bias,
                                                     _Float16* __restrict__ gi) {
  __shared__ _Float16 aSh[16 * APAD];
  int tid  = threadIdx.x;
  int wave = tid >> 5;
  int lane = tid & 31;
  int L = lane & 15, g = lane >> 4;
  long mbase = (long)blockIdx.x * 16;

  stage_A_async(aSh, act + mbase * EMB, tid);
  __syncthreads();

  v8f acc[6] = {};
#pragma unroll
  for (int kc = 0; kc < 8; ++kc) {
    int k1 = kc * 32 + 8 * g;           // ISA A/B frag K layout per lane group g
    v16h a = frag_cat(*(const h8*)(aSh + L * APAD + k1),
                      *(const h8*)(aSh + L * APAD + k1 + 16));
    h8 blo[6], bhi[6];
#pragma unroll
    for (int i = 0; i < 6; ++i) {       // batch all B loads first
      const _Float16* bp = w + (long)((wave * 6 + i) * 16 + L) * EMB + k1;
      blo[i] = *(const h8*)bp;
      bhi[i] = *(const h8*)(bp + 16);
    }
#pragma unroll
    for (int i = 0; i < 6; ++i) {
      acc[i] = __builtin_amdgcn_wmma_f32_16x16x32_f16(false, a, false,
                                                      frag_cat(blo[i], bhi[i]),
                                                      (short)0, acc[i], false, false);
    }
  }
#pragma unroll
  for (int i = 0; i < 6; ++i) {
    int j = (wave * 6 + i) * 16 + L;
    float bv = bias[j];
#pragma unroll
    for (int r = 0; r < 8; ++r) {       // C layout: vgpr r -> row r + 8g
      long m = mbase + r + 8 * g;
      gi[m * GATES + j] = (_Float16)(acc[i][r] + bv);
    }
  }
}

// ---------------------------------------------------------------------------
// One GRU timestep (serial critical path): gh = h @ W_hh^T via WMMA (A from
// async-staged LDS), gates via LDS, fused nonlinearity + masked update.
__global__ __launch_bounds__(256) void gruStep_kernel(const _Float16* __restrict__ gi,
                                                      const _Float16* __restrict__ whh,
                                                      const float* __restrict__ bhh,
                                                      const int* __restrict__ lengths,
                                                      float* __restrict__ h,
                                                      _Float16* __restrict__ hf,
                                                      _Float16* __restrict__ out16,
                                                      float* __restrict__ out32,
                                                      int t) {
  __shared__ _Float16 aSh[16 * APAD];   //  8448 B: h tile
  __shared__ float gl[16][772];         // 49408 B: gates, padded conflict-free
  int tid  = threadIdx.x;
  int wave = tid >> 5;
  int lane = tid & 31;
  int L = lane & 15, g = lane >> 4;
  int row0 = blockIdx.x * 16;

  stage_A_async(aSh, hf + (long)row0 * EMB, tid);
  __syncthreads();

  v8f acc[6] = {};
#pragma unroll
  for (int kc = 0; kc < 8; ++kc) {
    int k1 = kc * 32 + 8 * g;
    v16h a = frag_cat(*(const h8*)(aSh + L * APAD + k1),
                      *(const h8*)(aSh + L * APAD + k1 + 16));
    h8 blo[6], bhi[6];
#pragma unroll
    for (int i = 0; i < 6; ++i) {
      const _Float16* bp = whh + (long)((wave * 6 + i) * 16 + L) * EMB + k1;
      blo[i] = *(const h8*)bp;
      bhi[i] = *(const h8*)(bp + 16);
    }
#pragma unroll
    for (int i = 0; i < 6; ++i) {
      acc[i] = __builtin_amdgcn_wmma_f32_16x16x32_f16(false, a, false,
                                                      frag_cat(blo[i], bhi[i]),
                                                      (short)0, acc[i], false, false);
    }
  }
#pragma unroll
  for (int i = 0; i < 6; ++i) {
    int jc = (wave * 6 + i) * 16 + L;
#pragma unroll
    for (int r = 0; r < 8; ++r) gl[r + 8 * g][jc] = acc[i][r];
  }

  // Prefetch next timestep's gi slice for this block (global_prefetch_b8).
  if (t + 1 < TDIM) {
    int pr = tid >> 4, ps = tid & 15;   // 16 rows x 16 segments of 96B
    __builtin_prefetch(gi + ((long)(row0 + pr) * TDIM + (t + 1)) * GATES + ps * 48, 0, 3);
  }
  __syncthreads();

#pragma unroll
  for (int e = 0; e < 16; ++e) {
    int idx = tid + 256 * e;            // 16 rows x 256 channels
    int r = idx >> 8;
    int j = idx & 255;
    int b = row0 + r;
    long mrow = (long)b * TDIM + t;
    bool m = t < lengths[b];
    float gir = (float)gi[mrow * GATES + j];
    float giz = (float)gi[mrow * GATES + EMB + j];
    float gin = (float)gi[mrow * GATES + 2 * EMB + j];
    float ghr = gl[r][j]           + bhh[j];
    float ghz = gl[r][EMB + j]     + bhh[EMB + j];
    float ghn = gl[r][2 * EMB + j] + bhh[2 * EMB + j];
    float rr = sigmoidf_(gir + ghr);
    float zz = sigmoidf_(giz + ghz);
    float nn = tanhf(gin + rr * ghn);   // n = tanh(i_n + r * h_n)
    long hidx = (long)b * EMB + j;
    float hold = h[hidx];
    float hnew = (1.0f - zz) * nn + zz * hold;
    float hsel = m ? hnew : hold;
    h[hidx]  = hsel;
    hf[hidx] = (_Float16)hsel;
    long oi = mrow * EMB + j;
    if (out32) out32[oi] = m ? hnew : 0.0f;
    else       out16[oi] = m ? (_Float16)hnew : (_Float16)0.0f;
  }
}

// ---------------------------------------------------------------------------
__global__ void initH_kernel(const float* __restrict__ h0, float* __restrict__ h,
                             _Float16* __restrict__ hf, int n) {
  int i = blockIdx.x * blockDim.x + threadIdx.x;
  if (i < n) { float v = h0[i]; h[i] = v; hf[i] = (_Float16)v; }
}
__global__ void copyH_kernel(const float* __restrict__ h, float* __restrict__ dst, int n) {
  int i = blockIdx.x * blockDim.x + threadIdx.x;
  if (i < n) dst[i] = h[i];
}

// ---------------------------------------------------------------------------
extern "C" void kernel_launch(void* const* d_in, const int* in_sizes, int n_in,
                              void* d_out, int out_size, void* d_ws, size_t ws_size,
                              hipStream_t stream) {
  (void)in_sizes; (void)n_in; (void)out_size;
  const int*   x       = (const int*)d_in[0];
  const int*   lengths = (const int*)d_in[1];
  const float* hidden  = (const float*)d_in[2];
  const float* emb     = (const float*)d_in[3];
  const float* w_ih    = (const float*)d_in[4];
  const float* w_hh    = (const float*)d_in[5];
  const float* b_ih    = (const float*)d_in[6];
  const float* b_hh    = (const float*)d_in[7];
  float* out = (float*)d_out;

  // workspace layout (256B-aligned offsets), total ~86.25 MB
  char* ws = (char*)d_ws;
  _Float16* pooled16 = (_Float16*)(ws);              // 16,777,216 B  (B*T*E f16)
  _Float16* act16    = (_Float16*)(ws + 16777216);   // 16,777,216 B  (layer1 out f16)
  _Float16* gi16     = (_Float16*)(ws + 33554432);   // 50,331,648 B  (B*T*768 f16)
  float*    h        = (float*)   (ws + 83886080);   //    524,288 B  (B*E f32)
  _Float16* hf       = (_Float16*)(ws + 84410368);   //    262,144 B  (B*E f16)
  _Float16* wih16    = (_Float16*)(ws + 84672512);   //    786,432 B
  _Float16* whh16    = (_Float16*)(ws + 85458944);   //    786,432 B
  if (ws_size < 86245376u) return;

  const int WN = NLAY * GATES * EMB;                 // 393216 per matrix family
  wconv_kernel<<<(WN + 255) / 256, 256, 0, stream>>>(w_ih, w_hh, wih16, whh16, WN);
  pool_kernel<<<BDIM * TDIM, EMB, 0, stream>>>(x, emb, pooled16);

  for (int l = 0; l < NLAY; ++l) {
    const _Float16* actl = (l == 0) ? pooled16 : act16;
    initH_kernel<<<(BDIM * EMB + 255) / 256, 256, 0, stream>>>(
        hidden + (long)l * BDIM * EMB, h, hf, BDIM * EMB);
    // one block per 16-row M-tile: 32768/16 = 2048 blocks
    giGemm_kernel<<<2048, 256, 0, stream>>>(actl, wih16 + (long)l * GATES * EMB,
                                            b_ih + l * GATES, gi16);
    for (int t = 0; t < TDIM; ++t) {
      gruStep_kernel<<<BDIM / 16, 256, 0, stream>>>(
          gi16, whh16 + (long)l * GATES * EMB, b_hh + l * GATES, lengths, h, hf,
          (l == 0) ? act16 : (_Float16*)nullptr,
          (l == 0) ? (float*)nullptr : out, t);
    }
    copyH_kernel<<<(BDIM * EMB + 255) / 256, 256, 0, stream>>>(
        h, out + (long)BDIM * TDIM * EMB + (long)l * BDIM * EMB, BDIM * EMB);
  }
}
